// ODEEncoder_70566312673373
// MI455X (gfx1250) — compile-verified, matrix operand order checked by reading, not run on hardware
//
#include <hip/hip_runtime.h>
#include <hip/hip_bf16.h>

typedef __attribute__((ext_vector_type(16))) _Float16 v16h;
typedef __attribute__((ext_vector_type(8)))  _Float16 v8h;
typedef __attribute__((ext_vector_type(8)))  float    v8f;

#define B_SZ 256
#define T_SZ 1024
#define IN_F 64
#define Z_F  32
#define H_F  128
#define TM1  1023                 // T-1
#define NTILES (B_SZ * TM1 / 16)  // 16368

// Raw hardware log (v_log_f32 is log2): safe here because callers only pass
// arguments in [1,2] -- no denorm/inf edge cases, so skip libm's fixup code.
__device__ __forceinline__ float raw_ln(float x) {
#if __has_builtin(__builtin_amdgcn_logf)
  return 0.6931471805599453f * __builtin_amdgcn_logf(x);
#else
  return __logf(x);
#endif
}

// Branchless softplus: max(x,0) + ln(1 + exp(-|x|))
//   -> v_mul, v_exp_f32, v_add, v_log_f32, v_fma, v_max  (~6 ops)
__device__ __forceinline__ float softplus_f(float x) {
  const float e = __expf(-fabsf(x));
  return fmaxf(x, 0.f) + raw_ln(1.f + e);
}

__device__ __forceinline__ float fast_rcp(float x) {
#if __has_builtin(__builtin_amdgcn_rcpf)
  return __builtin_amdgcn_rcpf(x);
#else
  return 1.f / x;
#endif
}

__device__ __forceinline__ float fast_tanh(float x) {
#if __has_builtin(__builtin_amdgcn_tanhf)
  return __builtin_amdgcn_tanhf(x);
#elif __has_builtin(__builtin_amdgcn_tanh_f32)
  return __builtin_amdgcn_tanh_f32(x);
#else
  const float e = __expf(2.f * x);          // v_exp_f32
  return 1.f - 2.f * fast_rcp(e + 1.f);     // v_rcp_f32
#endif
}

__device__ __forceinline__ v8f wmma16(v16h a, v16h b, v8f c) {
  // D = A(16x32 f16) * B(32x16 f16) + C(16x16 f32)
  return __builtin_amdgcn_wmma_f32_16x16x32_f16(false, a, false, b, (short)0, c,
                                                false, false);
}

// ---------------------------------------------------------------------------
// Kernel 1: fused 2-layer tanh-RNN scan over reversed time.
// One wave32 per batch row; lane = hidden unit; h broadcast via shfl;
// weights staged in LDS with padded strides (65/33) to avoid bank conflicts.
// ---------------------------------------------------------------------------
__global__ __launch_bounds__(128) void rnn_scan_kernel(
    const float* __restrict__ x,
    const float* __restrict__ wih0, const float* __restrict__ whh0,
    const float* __restrict__ bih0, const float* __restrict__ bhh0,
    const float* __restrict__ wih1, const float* __restrict__ whh1,
    const float* __restrict__ bih1, const float* __restrict__ bhh1,
    float* __restrict__ ws_y, float* __restrict__ out) {
  __shared__ float sWih0[32 * 65];
  __shared__ float sWhh0[32 * 33];
  __shared__ float sWih1[32 * 33];
  __shared__ float sWhh1[32 * 33];
  __shared__ float sB0[32], sB1[32];

  const int tid = threadIdx.x;
  for (int i = tid; i < 32 * 64; i += 128)
    sWih0[(i >> 6) * 65 + (i & 63)] = wih0[i];
  for (int i = tid; i < 32 * 32; i += 128) {
    int r = i >> 5, c = i & 31;
    sWhh0[r * 33 + c] = whh0[i];
    sWih1[r * 33 + c] = wih1[i];
    sWhh1[r * 33 + c] = whh1[i];
  }
  if (tid < 32) {
    sB0[tid] = bih0[tid] + bhh0[tid];
    sB1[tid] = bih1[tid] + bhh1[tid];
  }
  __syncthreads();

  const int wave = tid >> 5;
  const int lane = tid & 31;
  const int b = blockIdx.x * 4 + wave;

  const float* xb  = x + (long)b * T_SZ * IN_F;
  const float* wr0 = sWih0 + lane * 65;
  const float* hr0 = sWhh0 + lane * 33;
  const float* ir1 = sWih1 + lane * 33;
  const float* hr1 = sWhh1 + lane * 33;
  const float bb0 = sB0[lane];
  const float bb1 = sB1[lane];

  float h0 = 0.f, h1 = 0.f;
  for (int tp = 0; tp < T_SZ; ++tp) {
    const int tt = (T_SZ - 1) - tp;           // time index into original x
    const float xlo = xb[tt * IN_F + lane];
    const float xhi = xb[tt * IN_F + 32 + lane];

    float acc0 = bb0;
#pragma unroll
    for (int j = 0; j < 32; ++j) {
      acc0 += wr0[j]      * __shfl(xlo, j, 32);
      acc0 += wr0[32 + j] * __shfl(xhi, j, 32);
      acc0 += hr0[j]      * __shfl(h0,  j, 32);
    }
    const float h0n = fast_tanh(acc0);

    float acc1 = bb1;
#pragma unroll
    for (int j = 0; j < 32; ++j) {
      acc1 += ir1[j] * __shfl(h0n, j, 32);
      acc1 += hr1[j] * __shfl(h1,  j, 32);
    }
    h1 = fast_tanh(acc1);
    h0 = h0n;

    if (tt >= 1)
      ws_y[((long)b * TM1 + (tt - 1)) * Z_F + lane] = h1;   // y0 seed
    if (tt == T_SZ - 1)
      out[((long)b * T_SZ + (T_SZ - 1)) * Z_F + lane] = h1; // last output row
  }
}

// ---------------------------------------------------------------------------
// Kernel 2: fully fused RK4 ODE integration (9 steps x 4 f-evals), WMMA f16
// with f32 accumulation.  3 waves/block, each wave grid-striding over 31
// 16-row tiles (weight-fragment fill amortized).  All LDS is static
// __shared__ (ds_* lowering).  The 16x32 staging buffer aliases the head of
// the 16x128 act buffer: the A0 fragment is consumed before layer-0's first
// overlapping store, and per-wave DS ops execute in order.
// ---------------------------------------------------------------------------
__global__ __launch_bounds__(96) void ode_rk4_fused(
    const float* __restrict__ yin, float* __restrict__ out,
    const float* __restrict__ w0, const float* __restrict__ b0,
    const float* __restrict__ w1, const float* __restrict__ b1,
    const float* __restrict__ fcw, const float* __restrict__ fcb) {
  __shared__ _Float16 sFragW0[8 * 512];    //  8 KB  B-frags, W0^T (32->128)
  __shared__ _Float16 sFragW1[32 * 512];   // 32 KB  B-frags, W1^T (128->128)
  __shared__ _Float16 sFragFC[8 * 512];    //  8 KB  B-frags, fc^T (128->32)
  __shared__ float    sBias[288];          //  b0 | b1 | fcb
  __shared__ _Float16 sAct[3 * 2048];      //  3 waves * 16x128 f16 (4 KB each)

  const int tid  = threadIdx.x;
  const int wave = tid >> 5;
  const int lane = tid & 31;

  // ---- pack weights (f32 -> f16) into per-lane WMMA B-fragments -----------
  for (int idx = tid; idx < 8 * 512; idx += 96) {     // W0^T
    int frag = idx >> 9, ln = (idx >> 4) & 31, e = idx & 15;
    int n = frag * 16 + (ln & 15);
    int k = (ln >> 4) * 8 + (e < 8 ? e : e + 8);
    sFragW0[idx] = (_Float16)w0[n * Z_F + k];
  }
  for (int idx = tid; idx < 32 * 512; idx += 96) {    // W1^T
    int frag = idx >> 9, ln = (idx >> 4) & 31, e = idx & 15;
    int nT = frag >> 2, kS = frag & 3;
    int n = nT * 16 + (ln & 15);
    int k = kS * 32 + (ln >> 4) * 8 + (e < 8 ? e : e + 8);
    sFragW1[idx] = (_Float16)w1[n * H_F + k];
  }
  for (int idx = tid; idx < 8 * 512; idx += 96) {     // fc^T
    int frag = idx >> 9, ln = (idx >> 4) & 31, e = idx & 15;
    int nT = frag >> 2, kS = frag & 3;
    int n = nT * 16 + (ln & 15);
    int k = kS * 32 + (ln >> 4) * 8 + (e < 8 ? e : e + 8);
    sFragFC[idx] = (_Float16)fcw[n * H_F + k];
  }
  for (int i = tid; i < 128; i += 96) {
    sBias[i]       = b0[i];
    sBias[128 + i] = b1[i];
  }
  if (tid < 32) sBias[256 + tid] = fcb[tid];
  __syncthreads();

  _Float16* actB = sAct + wave * 2048;  // 16x128 act buffer (layer0/layer1)
  _Float16* actA = actB;                // 16x32 staging aliases its head

  const int half = lane >> 4;     // C-layout half (rows 0-7 vs 8-15)
  const int colc = lane & 15;     // C/D column within a 16-wide tile
  const int am   = colc;          // A-fragment row for this lane
  const int akb  = half * 8;      // A-fragment K base for this lane

  // f(y): two 16x16 C-layout register tiles in, two out.
  auto evalF = [&](const v8f& a0, const v8f& a1, v8f& o0, v8f& o1) {
    // stage f's input as f16 (row-major 16x32) for the A-fragment transpose
#pragma unroll
    for (int r = 0; r < 8; ++r) {
      const int m = r + half * 8;
      actA[m * 32 + colc]      = (_Float16)a0[r];
      actA[m * 32 + 16 + colc] = (_Float16)a1[r];
    }
    v8h alo = *(const v8h*)(actA + am * 32 + akb);
    v8h ahi = *(const v8h*)(actA + am * 32 + akb + 16);
    const v16h A0 = __builtin_shufflevector(alo, ahi, 0, 1, 2, 3, 4, 5, 6, 7,
                                            8, 9, 10, 11, 12, 13, 14, 15);
    // layer 0: (16x32) @ W0^T -> softplus -> actB (16x128)
#pragma unroll
    for (int nT = 0; nT < 8; ++nT) {
      const v16h Bf = *(const v16h*)(sFragW0 + nT * 512 + lane * 16);
      v8f c = {};
      c = wmma16(A0, Bf, c);
#pragma unroll
      for (int r = 0; r < 8; ++r) {
        const int m = r + half * 8;
        const float v = softplus_f(c[r] + sBias[nT * 16 + colc]);
        actB[m * H_F + nT * 16 + colc] = (_Float16)v;
      }
    }
    // layer 1: (16x128) @ W1^T, all 8 output tiles accumulated in registers;
    // each A fragment loaded once and reused for 8 WMMAs.
    v8f cc[8];
#pragma unroll
    for (int nT = 0; nT < 8; ++nT) cc[nT] = (v8f){};
#pragma unroll
    for (int kS = 0; kS < 4; ++kS) {
      v8h lo = *(const v8h*)(actB + am * H_F + kS * 32 + akb);
      v8h hi = *(const v8h*)(actB + am * H_F + kS * 32 + akb + 16);
      const v16h A = __builtin_shufflevector(lo, hi, 0, 1, 2, 3, 4, 5, 6, 7,
                                             8, 9, 10, 11, 12, 13, 14, 15);
#pragma unroll
      for (int nT = 0; nT < 8; ++nT) {
        const v16h Bf =
            *(const v16h*)(sFragW1 + (nT * 4 + kS) * 512 + lane * 16);
        cc[nT] = wmma16(A, Bf, cc[nT]);
      }
    }
    // softplus + overwrite actB with layer-1 activations
#pragma unroll
    for (int nT = 0; nT < 8; ++nT) {
#pragma unroll
      for (int r = 0; r < 8; ++r) {
        const int m = r + half * 8;
        const float v = softplus_f(cc[nT][r] + sBias[128 + nT * 16 + colc]);
        actB[m * H_F + nT * 16 + colc] = (_Float16)v;
      }
    }
    // fc: (16x128) @ fc^T + b -> o (16x32 f32, C layout)
    v8f c0 = {}, c1 = {};
#pragma unroll
    for (int kS = 0; kS < 4; ++kS) {
      v8h lo = *(const v8h*)(actB + am * H_F + kS * 32 + akb);
      v8h hi = *(const v8h*)(actB + am * H_F + kS * 32 + akb + 16);
      const v16h A = __builtin_shufflevector(lo, hi, 0, 1, 2, 3, 4, 5, 6, 7,
                                             8, 9, 10, 11, 12, 13, 14, 15);
      const v16h Bf0 = *(const v16h*)(sFragFC + kS * 512 + lane * 16);
      const v16h Bf1 = *(const v16h*)(sFragFC + (4 + kS) * 512 + lane * 16);
      c0 = wmma16(A, Bf0, c0);
      c1 = wmma16(A, Bf1, c1);
    }
#pragma unroll
    for (int r = 0; r < 8; ++r) {
      o0[r] = c0[r] + sBias[256 + colc];
      o1[r] = c1[r] + sBias[256 + 16 + colc];
    }
  };

  const float hs = 0.1f / 10.f;  // DT / DISC
  const long wstride = (long)gridDim.x * 3;

  for (long tile = (long)blockIdx.x * 3 + wave; tile < NTILES; tile += wstride) {
    const long row0 = tile * 16;
    // ---- load y state (16 rows x 32 cols) into C-layout registers ---------
    v8f y0, y1;
#pragma unroll
    for (int r = 0; r < 8; ++r) {
      const int m = r + half * 8;
      const float* src = yin + (row0 + m) * Z_F;
      y0[r] = src[colc];
      y1[r] = src[16 + colc];
    }

#pragma unroll 1
    for (int s = 0; s < 9; ++s) {  // DISC - 1 RK4 steps
      v8f ka, kb, ta, tb, acc0, acc1;
      evalF(y0, y1, ka, kb);                         // k1
#pragma unroll
      for (int r = 0; r < 8; ++r) {
        acc0[r] = ka[r];
        acc1[r] = kb[r];
        ta[r] = y0[r] + 0.5f * hs * ka[r];
        tb[r] = y1[r] + 0.5f * hs * kb[r];
      }
      evalF(ta, tb, ka, kb);                         // k2
#pragma unroll
      for (int r = 0; r < 8; ++r) {
        acc0[r] += 2.f * ka[r];
        acc1[r] += 2.f * kb[r];
        ta[r] = y0[r] + 0.5f * hs * ka[r];
        tb[r] = y1[r] + 0.5f * hs * kb[r];
      }
      evalF(ta, tb, ka, kb);                         // k3
#pragma unroll
      for (int r = 0; r < 8; ++r) {
        acc0[r] += 2.f * ka[r];
        acc1[r] += 2.f * kb[r];
        ta[r] = y0[r] + hs * ka[r];
        tb[r] = y1[r] + hs * kb[r];
      }
      evalF(ta, tb, ka, kb);                         // k4
#pragma unroll
      for (int r = 0; r < 8; ++r) {
        y0[r] += (hs / 6.f) * (acc0[r] + ka[r]);
        y1[r] += (hs / 6.f) * (acc1[r] + kb[r]);
      }
    }

    // ---- scatter result: global row -> (batch b, time i), out[:, 0:T-1, :]
#pragma unroll
    for (int r = 0; r < 8; ++r) {
      const long gr = row0 + r + half * 8;
      const long bb = gr / TM1;
      const long ii = gr - bb * TM1;
      float* dst = out + (bb * T_SZ + ii) * Z_F;
      dst[colc]      = y0[r];
      dst[16 + colc] = y1[r];
    }
  }
}

extern "C" void kernel_launch(void* const* d_in, const int* in_sizes, int n_in,
                              void* d_out, int out_size, void* d_ws,
                              size_t ws_size, hipStream_t stream) {
  const float* x    = (const float*)d_in[0];
  // d_in[1] = seq_len (scalar), d_in[2] = h0 (zeros) -> unused
  const float* wih0 = (const float*)d_in[3];
  const float* whh0 = (const float*)d_in[4];
  const float* bih0 = (const float*)d_in[5];
  const float* bhh0 = (const float*)d_in[6];
  const float* wih1 = (const float*)d_in[7];
  const float* whh1 = (const float*)d_in[8];
  const float* bih1 = (const float*)d_in[9];
  const float* bhh1 = (const float*)d_in[10];
  const float* ow0  = (const float*)d_in[11];
  const float* ob0  = (const float*)d_in[12];
  const float* ow1  = (const float*)d_in[13];
  const float* ob1  = (const float*)d_in[14];
  const float* fcw  = (const float*)d_in[15];
  const float* fcb  = (const float*)d_in[16];

  float* out  = (float*)d_out;
  float* ws_y = (float*)d_ws;  // B * (T-1) * Z f32 = ~33.5 MB

  // Phase 1: sequential RNN scan (latency-bound; 1 wave per batch row)
  rnn_scan_kernel<<<B_SZ / 4, 128, 0, stream>>>(
      x, wih0, whh0, bih0, bhh0, wih1, whh1, bih1, bhh1, ws_y, out);

  // Phase 2: fused RK4 (compute-bound; WMMA f16 -> f32).
  // 176 blocks x 3 waves = 528 waves, each handling exactly 31 tiles.
  ode_rk4_fused<<<176, 96, 0, stream>>>(ws_y, out, ow0, ob0, ow1, ob1,
                                        fcw, fcb);
}
